// HGNN_Model_40346922778657
// MI455X (gfx1250) — matrix operand order repaired and checked by reading, not compiled, not run on hardware
//
#include <hip/hip_runtime.h>

typedef __bf16 v4bf  __attribute__((ext_vector_type(4)));
typedef __bf16 v8bf  __attribute__((ext_vector_type(8)));
typedef __bf16 v16bf __attribute__((ext_vector_type(16)));
typedef float  v8f   __attribute__((ext_vector_type(8)));
typedef int    v4i   __attribute__((ext_vector_type(4)));

#define TPB 256
#define BM  128
#define BN  128
#define BK  32
#define LDT 40   // padded LDS K-stride (bf16): 80B rows

static __device__ __forceinline__ v16bf cat8(v8bf lo, v8bf hi) {
  return __builtin_shufflevector(lo, hi, 0,1,2,3,4,5,6,7,8,9,10,11,12,13,14,15);
}
static __device__ __forceinline__ v8f wmma_bf16(v16bf a, v16bf b, v8f c) {
  return __builtin_amdgcn_wmma_f32_16x16x32_bf16(false, a, false, b, (short)0, c,
                                                 false, false);
}

// --------------------------- async global->LDS path ------------------------
#if defined(__AMDGCN__) && __has_builtin(__builtin_amdgcn_global_load_async_to_lds_b128)
#define HAVE_ASYNC_LDS 1
typedef __attribute__((address_space(1))) v4i g_v4i;
typedef __attribute__((address_space(3))) v4i l_v4i;
static __device__ __forceinline__ void async_cp16(void* lds, const void* gsrc) {
  __builtin_amdgcn_global_load_async_to_lds_b128((g_v4i*)gsrc, (l_v4i*)lds, 0, 0);
}
static __device__ __forceinline__ void async_wait0() {
#if __has_builtin(__builtin_amdgcn_s_wait_asynccnt)
  __builtin_amdgcn_s_wait_asynccnt(0);
#else
  asm volatile("s_wait_asynccnt 0x0" ::: "memory");
#endif
}
#else
#define HAVE_ASYNC_LDS 0
#endif

// ---------------------------------------------------------------------------
// G = DV^-1/2 Hs DE^-1 Hs^T DV^-1/2, sigmoid(H); bf16 [80][96], cols 80..95 = 0
// so the aggregation GEMM runs K = 96 in three 32-wide WMMA steps.
// ---------------------------------------------------------------------------
__global__ void gen_G_kernel(const float* __restrict__ H, __bf16* __restrict__ Gb) {
  __shared__ float Hs[80][81];
  __shared__ float dv2[80];
  __shared__ float invde[80];
  const int tid = threadIdx.x;
  for (int i = tid; i < 80 * 80; i += TPB) {
    float h = H[i];
    Hs[i / 80][i % 80] = 1.0f / (1.0f + __expf(-h));
  }
  __syncthreads();
  if (tid < 80) {
    float s = 0.f;
    for (int e = 0; e < 80; ++e) s += Hs[tid][e];
    dv2[tid] = rsqrtf(s);
  } else if (tid >= 128 && tid < 208) {
    const int e = tid - 128;
    float s = 0.f;
    for (int n = 0; n < 80; ++n) s += Hs[n][e];
    invde[e] = 1.0f / s;
  }
  __syncthreads();
  for (int idx = tid; idx < 80 * 96; idx += TPB) {
    const int m = idx / 96, n = idx % 96;
    float g = 0.f;
    if (n < 80) {
      float s = 0.f;
      for (int e = 0; e < 80; ++e) s += Hs[m][e] * invde[e] * Hs[n][e];
      g = s * dv2[m] * dv2[n];
    }
    Gb[idx] = (__bf16)g;
  }
}

// --------------------- f32 -> bf16 linear convert (X) ----------------------
__global__ void cvt_bf16_kernel(const float* __restrict__ src,
                                __bf16* __restrict__ dst, size_t n4) {
  const size_t i = (size_t)blockIdx.x * TPB + threadIdx.x;
  if (i < n4) {
    const float4 f = ((const float4*)src)[i];
    v4bf o = { (__bf16)f.x, (__bf16)f.y, (__bf16)f.z, (__bf16)f.w };
    ((v4bf*)dst)[i] = o;
  }
}

// ------------- f32 -> bf16 transposed convert: Wt[n][k] = W[k][n] ----------
__global__ void cvt_T_kernel(const float* __restrict__ W, __bf16* __restrict__ Wt,
                             int D) {
  __shared__ float ts[32][33];
  const int tx = threadIdx.x & 31, ty = threadIdx.x >> 5;   // 32 x 8
  const int k0 = blockIdx.x * 32, n0 = blockIdx.y * 32;
#pragma unroll
  for (int j = 0; j < 4; ++j)
    ts[ty + j * 8][tx] = W[(size_t)(k0 + ty + j * 8) * D + (n0 + tx)];
  __syncthreads();
#pragma unroll
  for (int j = 0; j < 4; ++j)
    Wt[(size_t)(n0 + ty + j * 8) * D + (k0 + tx)] = (__bf16)ts[tx][ty + j * 8];
}

// ---------------------------------------------------------------------------
// Ct[n][m] = sum_k A[m][k]*Wt[n][k] + bias[n]  (output stored transposed!)
// 128x128x32 tiles, 8 waves (4 along M x 2 along N), 2x4 WMMA tiles per wave.
// Both LDS tiles are identical straight copies (A row-major, Wt pre-transposed)
// -> async global->LDS when available, no staging VGPRs, no spills.
// Grid: (N/128, M/128).
// ---------------------------------------------------------------------------
static __device__ __forceinline__ void
mma_step(const __bf16 (*sa)[LDT], const __bf16 (*sb)[LDT],
         int wm, int wn, int frow, int kh, v8f acc[2][4]) {
  v16bf af[2], bfv[4];
#pragma unroll
  for (int mt = 0; mt < 2; ++mt) {
    const int row = wm * 32 + mt * 16 + frow;
    v8bf lo = *(const v8bf*)&sa[row][kh * 8];
    v8bf hi = *(const v8bf*)&sa[row][kh * 8 + 16];
    af[mt] = cat8(lo, hi);
  }
#pragma unroll
  for (int nt = 0; nt < 4; ++nt) {
    const int col = wn * 64 + nt * 16 + frow;
    v8bf lo = *(const v8bf*)&sb[col][kh * 8];
    v8bf hi = *(const v8bf*)&sb[col][kh * 8 + 16];
    bfv[nt] = cat8(lo, hi);
  }
#pragma unroll
  for (int mt = 0; mt < 2; ++mt)
#pragma unroll
    for (int nt = 0; nt < 4; ++nt)
      acc[mt][nt] = wmma_bf16(af[mt], bfv[nt], acc[mt][nt]);
}

__global__ void __launch_bounds__(TPB)
gemm_bf16_kernel(const __bf16* __restrict__ A, const __bf16* __restrict__ Bt,
                 const float* __restrict__ bias, __bf16* __restrict__ Ct,
                 int N, int K, int ldC) {
  __shared__ __align__(16) __bf16 sA[2][BM][LDT];
  __shared__ __align__(16) __bf16 sB[2][BN][LDT];

  const int tid  = threadIdx.x;
  const int m0   = blockIdx.y * BM;
  const int n0   = blockIdx.x * BN;
  const int lane = tid & 31;
  const int wv   = tid >> 5;
  const int wm   = wv & 3, wn = wv >> 2;
  const int frow = lane & 15, kh = lane >> 4;

  int r[2], c[2];
#pragma unroll
  for (int it = 0; it < 2; ++it) {
    const int idx = tid + it * TPB;
    r[it] = idx >> 2;  c[it] = (idx & 3) * 8;     // 128 rows x 32 k, 16B chunks
  }
  const __bf16* gA[2]; const __bf16* gB[2];
#pragma unroll
  for (int it = 0; it < 2; ++it) {
    gA[it] = A  + (size_t)(m0 + r[it]) * K + c[it];
    gB[it] = Bt + (size_t)(n0 + r[it]) * K + c[it];
  }
  const int nk = K / BK;

  v8f acc[2][4];
#pragma unroll
  for (int mt = 0; mt < 2; ++mt)
#pragma unroll
    for (int nt = 0; nt < 4; ++nt) acc[mt][nt] = (v8f)(0.0f);

#if HAVE_ASYNC_LDS
#pragma unroll
  for (int it = 0; it < 2; ++it) {
    async_cp16(&sA[0][r[it]][c[it]], gA[it]);
    async_cp16(&sB[0][r[it]][c[it]], gB[it]);
  }
  async_wait0();
  __syncthreads();
  for (int kt = 0; kt < nk; ++kt) {
    const int buf = kt & 1;
    if (kt + 1 < nk) {                 // issue next-tile copies before compute
      const int k0 = (kt + 1) * BK;
      const int nb = buf ^ 1;
#pragma unroll
      for (int it = 0; it < 2; ++it) {
        async_cp16(&sA[nb][r[it]][c[it]], gA[it] + k0);
        async_cp16(&sB[nb][r[it]][c[it]], gB[it] + k0);
      }
    }
    mma_step(sA[buf], sB[buf], wm, wn, frow, kh, acc);
    if (kt + 1 < nk) { async_wait0(); __syncthreads(); }
  }
#else
  {
    uint4 ra[2], rb[2];
#pragma unroll
    for (int it = 0; it < 2; ++it) {
      ra[it] = *(const uint4*)gA[it];
      rb[it] = *(const uint4*)gB[it];
    }
#pragma unroll
    for (int it = 0; it < 2; ++it) {
      *(uint4*)&sA[0][r[it]][c[it]] = ra[it];
      *(uint4*)&sB[0][r[it]][c[it]] = rb[it];
    }
    __syncthreads();
    for (int kt = 0; kt < nk; ++kt) {
      const int buf = kt & 1;
      if (kt + 1 < nk) {
        const int k0 = (kt + 1) * BK;
#pragma unroll
        for (int it = 0; it < 2; ++it) {
          ra[it] = *(const uint4*)(gA[it] + k0);
          rb[it] = *(const uint4*)(gB[it] + k0);
        }
        if (kt + 2 < nk) {
          const int kp = (kt + 2) * BK;
          __builtin_prefetch(gA[0] + kp, 0, 0);
          __builtin_prefetch(gB[0] + kp, 0, 0);
        }
      }
      mma_step(sA[buf], sB[buf], wm, wn, frow, kh, acc);
      if (kt + 1 < nk) {
        const int nb = buf ^ 1;
#pragma unroll
        for (int it = 0; it < 2; ++it) {
          *(uint4*)&sA[nb][r[it]][c[it]] = ra[it];
          *(uint4*)&sB[nb][r[it]][c[it]] = rb[it];
        }
        __syncthreads();
      }
    }
  }
#endif

  // transposed epilogue: one 16B v8bf store per accumulator tile
#pragma unroll
  for (int mt = 0; mt < 2; ++mt) {
    const int mrow = m0 + wm * 32 + mt * 16 + kh * 8;
#pragma unroll
    for (int nt = 0; nt < 4; ++nt) {
      const int n  = n0 + wn * 64 + nt * 16 + frow;
      const float bv = bias[n];
      const v8f t = acc[mt][nt];
      v8bf o;
#pragma unroll
      for (int i = 0; i < 8; ++i) o[i] = (__bf16)(t[i] + bv);
      *(v8bf*)(Ct + (size_t)n * ldC + mrow) = o;
    }
  }
}

// ---------------------------------------------------------------------------
// Per-batch aggregation out[b] = G(80x80, K-padded 96) @ Y[b](80xN).
// Yt is the transposed GEMM output [N][ldY]: per feature row n, the node index
// k is contiguous -> WMMA B-fragments load straight from global (L2-resident),
// no LDS staging, no in-loop barriers. G lives in LDS, reused by all waves.
// Grid: (N/128, B). Writes bf16 Z (row-major, ld=N) or strided f32 into d_out.
// ---------------------------------------------------------------------------
__global__ void __launch_bounds__(TPB)
agg_wmma_kernel(const __bf16* __restrict__ Gb, const __bf16* __restrict__ Yt,
                int ldY, int N, __bf16* __restrict__ Zb, float* __restrict__ Fout,
                int ldF, int colOfs, int doRelu) {
  __shared__ __align__(16) __bf16 Gs[80][104];
  const int tid  = threadIdx.x;
  const int b    = blockIdx.y;
  const int n0   = blockIdx.x * BN;
  const int lane = tid & 31;
  const int wv   = tid >> 5;            // each wave owns one 16-wide N tile
  const int frow = lane & 15, kh = lane >> 4;

  for (int idx = tid; idx < 80 * 96; idx += TPB)
    Gs[idx / 96][idx % 96] = Gb[idx];
  __syncthreads();

  const int n = n0 + wv * 16 + frow;
  const __bf16* yp = Yt + (size_t)n * ldY + (size_t)b * 80 + kh * 16;

  v8f acc[5];
#pragma unroll
  for (int mt = 0; mt < 5; ++mt) acc[mt] = (v8f)(0.0f);

#pragma unroll
  for (int kt = 0; kt < 3; ++kt) {
    const int k0 = kt * BK;
    v8bf lo = *(const v8bf*)(yp + k0);
    v8bf hi = *(const v8bf*)(yp + k0 + 8);
    const v16bf bfrag = cat8(lo, hi);
#pragma unroll
    for (int mt = 0; mt < 5; ++mt) {
      const int row = mt * 16 + frow;
      v8bf alo = *(const v8bf*)&Gs[row][k0 + kh * 16];
      v8bf ahi = *(const v8bf*)&Gs[row][k0 + kh * 16 + 8];
      acc[mt] = wmma_bf16(cat8(alo, ahi), bfrag, acc[mt]);
    }
  }

#pragma unroll
  for (int mt = 0; mt < 5; ++mt) {
    const int mrow = mt * 16 + kh * 8;
#pragma unroll
    for (int i = 0; i < 8; ++i) {
      float v = acc[mt][i];
      if (doRelu) v = v > 0.0f ? v : 0.0f;
      const size_t rrow = (size_t)b * 80 + mrow + i;
      if (Fout) Fout[rrow * (size_t)ldF + colOfs + n] = v;
      else      Zb[rrow * (size_t)N + n] = (__bf16)v;
    }
  }
}

// ---------------------------------------------------------------------------
// Host-side pipeline
// ---------------------------------------------------------------------------
static void run_stage(const float* X, const float* w1, const float* b1,
                      const float* w2, const float* b2, int D, int colOfs,
                      __bf16* Gb, __bf16* W1t, __bf16* W2t,
                      __bf16* bufA, __bf16* bufYt, float* out,
                      hipStream_t stream) {
  const int M = 10240;                      // B * N = 128 * 80
  const int M2 = 10256;                     // + 16 slack cols (zero G columns)
  const size_t nX4 = (size_t)M * D / 4;
  cvt_bf16_kernel<<<(unsigned)((nX4 + TPB - 1) / TPB), TPB, 0, stream>>>(X, bufA, nX4);
  dim3 tg(D / 32, D / 32);
  cvt_T_kernel<<<tg, TPB, 0, stream>>>(w1, W1t, D);
  cvt_T_kernel<<<tg, TPB, 0, stream>>>(w2, W2t, D);

  dim3 gg(D / BN, M / BM);
  dim3 ag(D / BN, 128);
  // Y1t = (X@W1+b1)^T ; Z = relu(G@Y1) ; Y2t = (Z@W2+b2)^T ; out = G@Y2
  gemm_bf16_kernel<<<gg, TPB, 0, stream>>>(bufA, W1t, b1, bufYt, D, D, M2);
  agg_wmma_kernel<<<ag, TPB, 0, stream>>>(Gb, bufYt, M2, D, bufA, nullptr, 0, 0, 1);
  gemm_bf16_kernel<<<gg, TPB, 0, stream>>>(bufA, W2t, b2, bufYt, D, D, M2);
  agg_wmma_kernel<<<ag, TPB, 0, stream>>>(Gb, bufYt, M2, D, nullptr, out, 3072, colOfs, 0);
}

extern "C" void kernel_launch(void* const* d_in, const int* in_sizes, int n_in,
                              void* d_out, int out_size, void* d_ws, size_t ws_size,
                              hipStream_t stream) {
  (void)in_sizes; (void)n_in; (void)out_size; (void)ws_size;
  const float* x3  = (const float*)d_in[0];
  const float* x4  = (const float*)d_in[1];
  const float* H   = (const float*)d_in[2];
  const float* w31 = (const float*)d_in[3];
  const float* b31 = (const float*)d_in[4];
  const float* w32 = (const float*)d_in[5];
  const float* b32 = (const float*)d_in[6];
  const float* w41 = (const float*)d_in[7];
  const float* b41 = (const float*)d_in[8];
  const float* w42 = (const float*)d_in[9];
  const float* b42 = (const float*)d_in[10];
  float* out = (float*)d_out;

  char* p = (char*)d_ws;
  __bf16* Gb    = (__bf16*)p;  p += 32768;                       // 80x96 bf16 + pad
  __bf16* W1t   = (__bf16*)p;  p += (size_t)2048 * 2048 * 2;     // 8 MB
  __bf16* W2t   = (__bf16*)p;  p += (size_t)2048 * 2048 * 2;     // 8 MB
  __bf16* bufA  = (__bf16*)p;  p += (size_t)10240 * 2048 * 2;    // X, then Z (40 MB)
  __bf16* bufYt = (__bf16*)p;                                    // [D][10256] bf16

  gen_G_kernel<<<1, TPB, 0, stream>>>(H, Gb);

  run_stage(x3, w31, b31, w32, b32, 1024, 0,    Gb, W1t, W2t, bufA, bufYt, out, stream);
  run_stage(x4, w41, b41, w42, b42, 2048, 1024, Gb, W1t, W2t, bufA, bufYt, out, stream);
}
